// Model_69681549411002
// MI455X (gfx1250) — compile-verified, matrix operand order checked by reading, not run on hardware
//
#include <hip/hip_runtime.h>

// ---------------------------------------------------------------------------
// MI455X (gfx1250) grouped-expert pipeline.
// L=4, B=4096, H=1024, I=4096.
// Grouped (per-label) GEMM chain, bf16 WMMA with f32 accumulate.
// Block tile 128Mx128N, 8 waves (4x2), wave tile 32Mx64N (2x4 16x16 tiles),
// K-step 32, LDS double-buffered (1 barrier / K-step).
// A-tiles (bf16 activations) staged by the Tensor Data Mover (TDM) with
// LDS padding (64B rows + 16B pad -> pitch 80B) when the builtin exists.
// ---------------------------------------------------------------------------

#define LNG 4
#define BSZ 4096
#define HID 1024
#define INTER 4096

typedef __attribute__((ext_vector_type(16))) __bf16 v16bf;
typedef __attribute__((ext_vector_type(8)))  float  v8f;
typedef __attribute__((ext_vector_type(4)))  unsigned int v4u;
typedef __attribute__((ext_vector_type(8)))  int v8i;
typedef __attribute__((ext_vector_type(4)))  int v4i;

#if __has_builtin(__builtin_amdgcn_tensor_load_to_lds) && \
    __has_builtin(__builtin_amdgcn_s_wait_tensorcnt)
#define TDM_AVAIL 1
#else
#define TDM_AVAIL 0
#endif

__device__ __forceinline__ unsigned short f2bf(float f) {
    unsigned int u = __float_as_uint(f);
    u += 0x7FFFu + ((u >> 16) & 1u);          // round-to-nearest-even
    return (unsigned short)(u >> 16);
}

#if __has_builtin(__builtin_amdgcn_cvt_pk_bf16_f32)
typedef __attribute__((ext_vector_type(2))) __bf16 v2bf;
__device__ __forceinline__ unsigned int pack2(float lo, float hi) {
    v2bf v = __builtin_amdgcn_cvt_pk_bf16_f32(lo, hi);
    return __builtin_bit_cast(unsigned int, v);
}
#else
__device__ __forceinline__ unsigned int pack2(float lo, float hi) {
    return (unsigned int)f2bf(lo) | ((unsigned int)f2bf(hi) << 16);
}
#endif

__device__ __forceinline__ v16bf load_frag(const unsigned short* p) {
    // 32 bytes = 16 bf16; 16B-aligned => two b128 LDS reads
    union { uint4 u[2]; v16bf v; } x;
    const uint4* q = (const uint4*)p;
    x.u[0] = q[0];
    x.u[1] = q[1];
    return x.v;
}
__device__ __forceinline__ float wred(float v) {
#pragma unroll
    for (int o = 16; o > 0; o >>= 1) v += __shfl_xor(v, o, 32);
    return v;
}

#if TDM_AVAIL
// TDM: load a 2D tile [rows x 64 bytes] (row stride = strideBytes) into LDS
// at ldsOff, inserting 16B padding after each 64B row (pitch 80B = LDA*2).
// rowsAvail < 128 => OOB rows are zero-filled by the TDM.
__device__ __forceinline__ void tdm_load_a(const void* gsrc, unsigned int ldsOff,
                                           int rowsAvail, unsigned int strideBytes)
{
    unsigned long long ga = (unsigned long long)(size_t)gsrc;
    v4u g0;
    g0[0] = 1u;                                          // count=1 (valid D#)
    g0[1] = ldsOff;                                      // lds_addr
    g0[2] = (unsigned int)(ga & 0xFFFFFFFFull);          // global_addr[31:0]
    g0[3] = (unsigned int)((ga >> 32) & 0x1FFFFFFull)    // global_addr[56:32]
          | (2u << 30);                                  // type=2 ("image")
    v8i g1;
    g1[0] = (int)((1u << 20) | (3u << 22) | (3u << 25)); // pad_en, every 16 dw, pad 4 dw
    g1[1] = (int)(64u << 16);                            // tensor_dim0 = 64 (1B elems)
    g1[2] = (int)(((unsigned int)rowsAvail & 0xFFFFu) << 16); // tensor_dim1 lo16
    g1[3] = (int)((((unsigned int)rowsAvail >> 16) & 0xFFFFu) | (64u << 16)); // dim1 hi | tile_dim0=64
    g1[4] = (int)128u;                                   // tile_dim1=128, tile_dim2=0
    g1[5] = (int)strideBytes;                            // tensor_dim0_stride lo32
    g1[6] = 0;                                           // stride hi / dim1_stride lo
    g1[7] = 0;
    v4i g2 = {0, 0, 0, 0};
    v4i g3 = {0, 0, 0, 0};
    v8i g4 = {0, 0, 0, 0, 0, 0, 0, 0};
    __builtin_amdgcn_tensor_load_to_lds(g0, g1, g2, g3, g4, 0);
}
#endif

// ---------------------------------------------------------------------------
// Setup: group tokens by label
// ---------------------------------------------------------------------------
__global__ void init_kernel(float* out, int* counts) {
    int t = threadIdx.x;
    if (t < 3) out[t] = 0.0f;
    if (t < LNG) counts[t] = 0;
}
__global__ void count_kernel(const int* __restrict__ labels, int* counts) {
    int b = blockIdx.x * 256 + threadIdx.x;
    if (b < BSZ) atomicAdd(&counts[labels[b]], 1);
}
__global__ void offs_kernel(const int* counts, int* goff, int* gcnt, int* cursor) {
    if (threadIdx.x == 0) {
        int s = 0;
        for (int i = 0; i < LNG; ++i) {
            goff[i] = s; gcnt[i] = counts[i]; cursor[i] = s; s += counts[i];
        }
    }
}
__global__ void scatter_kernel(const int* __restrict__ labels, int* cursor, int* order) {
    int b = blockIdx.x * 256 + threadIdx.x;
    if (b < BSZ) {
        int pos = atomicAdd(&cursor[labels[b]], 1);
        order[pos] = b;
    }
}

// ---------------------------------------------------------------------------
// Contrastive loss: one wave (32 lanes) per token
// ---------------------------------------------------------------------------
__global__ __launch_bounds__(256) void contrastive_kernel(
    const float* __restrict__ a, const float* __restrict__ p,
    const float* __restrict__ n, float* out)
{
    const int wv = threadIdx.x >> 5, lane = threadIdx.x & 31;
    const int b = blockIdx.x * 8 + wv;
    const float* ar = a + (size_t)b * HID;
    const float* pr = p + (size_t)b * HID;
    const float* nr = n + (size_t)b * HID;
    float ps = 0.f, ns = 0.f;
#pragma unroll
    for (int i = 0; i < 8; ++i) {
        int h = lane * 4 + i * 128;
        float4 av = *(const float4*)(ar + h);
        float4 pv = *(const float4*)(pr + h);
        float4 nv = *(const float4*)(nr + h);
        ps += av.x * pv.x + av.y * pv.y + av.z * pv.z + av.w * pv.w;
        ns += av.x * nv.x + av.y * nv.y + av.z * nv.z + av.w * nv.w;
    }
    ps = wred(ps); ns = wred(ns);
    if (lane == 0) {
        float p0 = 1.0f / (1.0f + __expf(ns - ps));   // softmax([ps,ns])[0]
        atomicAdd(out, -__logf(p0 + 1e-10f) * (1.0f / BSZ));
    }
}

// ---------------------------------------------------------------------------
// Grouped GEMM: out[M,N] = act(Ag[M,K] @ W[slice][K,N] + bias[slice][N])
// EPI==0: relu -> bf16 out.  EPI==1: bias -> f32 out (+optional bf16 copy).
// ---------------------------------------------------------------------------
template <bool GATHER, int EPI>
__global__ __launch_bounds__(256) void gemm_bf16_kernel(
    const float* __restrict__ Af, const unsigned short* __restrict__ Ab,
    const int* __restrict__ order, const int* __restrict__ goff,
    const int* __restrict__ gcnt, const float* __restrict__ W,
    const float* __restrict__ bias, float* __restrict__ outF,
    unsigned short* __restrict__ outB, int K, int N, int cyc)
{
    constexpr int LDA = 40, LDB = 40;   // ushort strides (16B-aligned frags)
    constexpr bool TDM_A = (!GATHER) && (TDM_AVAIL != 0);
    __shared__ alignas(16) unsigned short lds_a[2][128 * LDA];
    __shared__ alignas(16) unsigned short lds_b[2][128 * LDB];

    const int g   = blockIdx.z;
    const int cnt = gcnt[g];
    const int mbase = blockIdx.y * 128;
    if (mbase >= cnt) return;
    const int go  = goff[g];
    const int tgt = (g < LNG - 1) ? (LNG - 1) : (LNG - 2);
    const int slice = cyc ? (tgt * LNG + g) : (g * LNG + tgt);
    const float* Wg = W + (size_t)slice * K * N;
    const float* bg = bias + (size_t)slice * N;
    const int nbase = blockIdx.x * 128;

    const int tid  = threadIdx.x;
    const int wv   = tid >> 5, lane = tid & 31, hf = lane >> 4, lr = lane & 15;
    const int mw   = wv & 3;            // M sub-block (32 rows each)
    const int nw   = wv >> 2;           // N sub-block (64 cols each)

    v8f acc[2][4] = {};

    // ---- A staging map (register path): row ar, k-half acb ----
    const int ar  = tid >> 1;              // 0..127
    const int acb = (tid & 1) << 4;        // 0 or 16
    int gm = mbase + ar;
    if (gm >= cnt) gm = cnt - 1;           // pad-row clamp (masked on store)
    const int srow = GATHER ? order[go + gm] : (go + gm);
    // ---- W staging map: k-pair (ke,ke+1) x 4 n, 2 reps (k +16) ----
    const int ke  = (tid >> 5) << 1;       // 0,2,..,14
    const int wn4 = (tid & 31) << 2;       // 0..124

    // staging registers
    uint4 aS0 = {0, 0, 0, 0}, aS1 = {0, 0, 0, 0};
    unsigned int wS[8];

    // ---- prologue loads (k0 = 0) ----
    if constexpr (TDM_A) {
#if TDM_AVAIL
        if (wv == 0) {
            tdm_load_a(Ab + (size_t)(go + mbase) * K,
                       (unsigned int)(size_t)&lds_a[0][0],
                       cnt - mbase, (unsigned int)K * 2u);
        }
#endif
    } else if constexpr (GATHER) {
        const float* src = Af + (size_t)srow * K + acb;
        float4 f0 = *(const float4*)(src + 0);
        float4 f1 = *(const float4*)(src + 4);
        float4 f2 = *(const float4*)(src + 8);
        float4 f3 = *(const float4*)(src + 12);
        aS0.x = pack2(f0.x, f0.y); aS0.y = pack2(f0.z, f0.w);
        aS0.z = pack2(f1.x, f1.y); aS0.w = pack2(f1.z, f1.w);
        aS1.x = pack2(f2.x, f2.y); aS1.y = pack2(f2.z, f2.w);
        aS1.z = pack2(f3.x, f3.y); aS1.w = pack2(f3.z, f3.w);
    } else {
        const uint4* src = (const uint4*)(Ab + (size_t)srow * K + acb);
        aS0 = src[0]; aS1 = src[1];
    }
#pragma unroll
    for (int rep = 0; rep < 2; ++rep) {
        const int kr = ke + rep * 16;
        const float* w0 = Wg + (size_t)kr * N + nbase + wn4;
        float4 flo = *(const float4*)w0;
        float4 fhi = *(const float4*)(w0 + N);
        wS[rep * 4 + 0] = pack2(flo.x, fhi.x);
        wS[rep * 4 + 1] = pack2(flo.y, fhi.y);
        wS[rep * 4 + 2] = pack2(flo.z, fhi.z);
        wS[rep * 4 + 3] = pack2(flo.w, fhi.w);
    }

    int cur = 0;
    for (int k0 = 0; k0 < K; k0 += 32) {
        // ---- commit staged W (and A on register path) to buffer `cur` ----
        if constexpr (!TDM_A) {
            uint4* dst = (uint4*)&lds_a[cur][ar * LDA + acb];
            dst[0] = aS0; dst[1] = aS1;
        }
#pragma unroll
        for (int rep = 0; rep < 2; ++rep) {
            const int kr = ke + rep * 16;
#pragma unroll
            for (int j = 0; j < 4; ++j) {
                *(unsigned int*)&lds_b[cur][(wn4 + j) * LDB + kr] = wS[rep * 4 + j];
            }
        }
#if TDM_AVAIL
        if constexpr (TDM_A) {
            if (wv == 0) __builtin_amdgcn_s_wait_tensorcnt((short)0);  // A[cur] landed
        }
#endif
        __syncthreads();

        // ---- issue next tile's loads (overlap with WMMA burst) ----
        const bool more = (k0 + 32 < K);
        if (more) {
            const int kn = k0 + 32;
            if constexpr (TDM_A) {
#if TDM_AVAIL
                if (wv == 0) {
                    tdm_load_a(Ab + (size_t)(go + mbase) * K + kn,
                               (unsigned int)(size_t)&lds_a[cur ^ 1][0],
                               cnt - mbase, (unsigned int)K * 2u);
                }
#endif
            } else if constexpr (GATHER) {
                const float* src = Af + (size_t)srow * K + kn + acb;
                float4 f0 = *(const float4*)(src + 0);
                float4 f1 = *(const float4*)(src + 4);
                float4 f2 = *(const float4*)(src + 8);
                float4 f3 = *(const float4*)(src + 12);
                aS0.x = pack2(f0.x, f0.y); aS0.y = pack2(f0.z, f0.w);
                aS0.z = pack2(f1.x, f1.y); aS0.w = pack2(f1.z, f1.w);
                aS1.x = pack2(f2.x, f2.y); aS1.y = pack2(f2.z, f2.w);
                aS1.z = pack2(f3.x, f3.y); aS1.w = pack2(f3.z, f3.w);
            } else {
                const uint4* src = (const uint4*)(Ab + (size_t)srow * K + kn + acb);
                aS0 = src[0]; aS1 = src[1];
            }
#pragma unroll
            for (int rep = 0; rep < 2; ++rep) {
                const int kr = ke + rep * 16;
                const float* w0 = Wg + (size_t)(kn + kr) * N + nbase + wn4;
                float4 flo = *(const float4*)w0;
                float4 fhi = *(const float4*)(w0 + N);
                wS[rep * 4 + 0] = pack2(flo.x, fhi.x);
                wS[rep * 4 + 1] = pack2(flo.y, fhi.y);
                wS[rep * 4 + 2] = pack2(flo.z, fhi.z);
                wS[rep * 4 + 3] = pack2(flo.w, fhi.w);
            }
        }

        // ---- WMMA burst: 2 A-frags x 4 B-frags = 8 wmma ----
        v16bf af0 = load_frag(&lds_a[cur][(mw * 32 +      lr) * LDA + hf * 16]);
        v16bf af1 = load_frag(&lds_a[cur][(mw * 32 + 16 + lr) * LDA + hf * 16]);
#pragma unroll
        for (int t = 0; t < 4; ++t) {
            v16bf bfr = load_frag(&lds_b[cur][(nw * 64 + t * 16 + lr) * LDB + hf * 16]);
            acc[0][t] = __builtin_amdgcn_wmma_f32_16x16x32_bf16(
                false, af0, false, bfr, (short)0, acc[0][t], false, false);
            acc[1][t] = __builtin_amdgcn_wmma_f32_16x16x32_bf16(
                false, af1, false, bfr, (short)0, acc[1][t], false, false);
        }
        cur ^= 1;
    }

    // ---- epilogue: C layout lane(lr)=N, VGPR r => M = 8*hf + r ----
#pragma unroll
    for (int mi = 0; mi < 2; ++mi) {
#pragma unroll
        for (int t = 0; t < 4; ++t) {
            union { v8f v; float f[8]; } u; u.v = acc[mi][t];
            const int n  = nbase + nw * 64 + t * 16 + lr;
            const float bv = bg[n];
#pragma unroll
            for (int r = 0; r < 8; ++r) {
                const int m = mbase + mw * 32 + mi * 16 + hf * 8 + r;
                if (m < cnt) {
                    float v = u.f[r] + bv;
                    const size_t idx = (size_t)(go + m) * N + n;
                    if (EPI == 0) {
                        outB[idx] = f2bf(fmaxf(v, 0.0f));
                    } else {
                        outF[idx] = v;
                        if (outB) outB[idx] = f2bf(v);
                    }
                }
            }
        }
    }
}

// ---------------------------------------------------------------------------
// Epilogue: per grouped position p -> token b = order[p]
// ---------------------------------------------------------------------------
__global__ __launch_bounds__(256) void epilogue_kernel(
    const float* __restrict__ anchor, const float* __restrict__ Tf,
    const float* __restrict__ Cf, const int* __restrict__ order,
    const float* __restrict__ Wd, const float* __restrict__ bd,
    const int* __restrict__ labels, float* __restrict__ out)
{
    const int p   = blockIdx.x;
    const int tid = threadIdx.x;
    const int lane = tid & 31;
    const int b   = order[p];
    const int lab = labels[b];

    __shared__ float sT, sC, sCY, sLG[LNG];
    if (tid == 0) { sT = 0.f; sC = 0.f; sCY = 0.f; }
    if (tid < LNG) sLG[tid] = 0.f;
    __syncthreads();

    const int h0 = tid * 4;
    float4 t4 = *(const float4*)(Tf + (size_t)p * HID + h0);
    float4 c4 = *(const float4*)(Cf + (size_t)p * HID + h0);
    float4 a4 = *(const float4*)(anchor + (size_t)b * HID + h0);
    float ta[4] = { t4.x, t4.y, t4.z, t4.w };
    float ca[4] = { c4.x, c4.y, c4.z, c4.w };
    float aa[4] = { a4.x, a4.y, a4.z, a4.w };

    float ssT = ta[0]*ta[0] + ta[1]*ta[1] + ta[2]*ta[2] + ta[3]*ta[3];
    float ssC = ca[0]*ca[0] + ca[1]*ca[1] + ca[2]*ca[2] + ca[3]*ca[3];
    ssT = wred(ssT); ssC = wred(ssC);
    if (lane == 0) { atomicAdd(&sT, ssT); atomicAdd(&sC, ssC); }
    __syncthreads();

    const float nT = fmaxf(sqrtf(sT), 1e-12f);
    const float nC = fmaxf(sqrtf(sC), 1e-12f);

    float lg[LNG] = {0.f, 0.f, 0.f, 0.f};
    float cy = 0.f;
#pragma unroll
    for (int j = 0; j < 4; ++j) {
        const int h = h0 + j;
        const float nt = ta[j] / nT;
#pragma unroll
        for (int l = 0; l < LNG; ++l) lg[l] += nt * Wd[h * LNG + l];
        cy += fabsf(aa[j] - ca[j] / nC);
        out[3 + (size_t)b * HID + h] = aa[j];     // anchor passthrough
    }
#pragma unroll
    for (int l = 0; l < LNG; ++l) lg[l] = wred(lg[l]);
    cy = wred(cy);
    if (lane == 0) {
#pragma unroll
        for (int l = 0; l < LNG; ++l) atomicAdd(&sLG[l], lg[l]);
        atomicAdd(&sCY, cy);
    }
    __syncthreads();

    if (tid == 0) {
        float z[LNG];
        float mx = -3.4e38f;
#pragma unroll
        for (int l = 0; l < LNG; ++l) { z[l] = sLG[l] + bd[l]; mx = fmaxf(mx, z[l]); }
        float se = 0.f;
#pragma unroll
        for (int l = 0; l < LNG; ++l) se += __expf(z[l] - mx);
        const float logp = z[lab] - mx - __logf(se);
        atomicAdd(&out[1], -logp * (1.0f / BSZ));
        atomicAdd(&out[2], sCY * (1.0f / ((float)BSZ * (float)HID)));
    }
}

// ---------------------------------------------------------------------------
extern "C" void kernel_launch(void* const* d_in, const int* in_sizes, int n_in,
                              void* d_out, int out_size, void* d_ws, size_t ws_size,
                              hipStream_t stream) {
    const float* anchor   = (const float*)d_in[0];
    const float* positive = (const float*)d_in[1];
    const float* negative = (const float*)d_in[2];
    const float* W1 = (const float*)d_in[3];
    const float* b1 = (const float*)d_in[4];
    const float* W2 = (const float*)d_in[5];
    const float* b2 = (const float*)d_in[6];
    const float* Wd = (const float*)d_in[7];
    const float* bd = (const float*)d_in[8];
    const int* labels = (const int*)d_in[9];
    float* out = (float*)d_out;

    char* ws = (char*)d_ws;
    int* counts = (int*)(ws + 0);
    int* goff   = (int*)(ws + 16);
    int* gcnt   = (int*)(ws + 32);
    int* cursor = (int*)(ws + 48);
    int* order  = (int*)(ws + 64);                         // 16 KB
    unsigned short* Hb = (unsigned short*)(ws + 32768);    // B*I bf16 = 32 MB (reused for HC)
    float*          Tf = (float*)(ws + 33587200);          // B*H f32  = 16 MB
    unsigned short* Tb = (unsigned short*)(ws + 50364416); // B*H bf16 =  8 MB
    float*          Cf = (float*)(ws + 58753024);          // B*H f32  = 16 MB

    init_kernel<<<1, 64, 0, stream>>>(out, counts);
    count_kernel<<<BSZ / 256, 256, 0, stream>>>(labels, counts);
    offs_kernel<<<1, 32, 0, stream>>>(counts, goff, gcnt, cursor);
    scatter_kernel<<<BSZ / 256, 256, 0, stream>>>(labels, cursor, order);
    contrastive_kernel<<<BSZ / 8, 256, 0, stream>>>(anchor, positive, negative, out);

    // H = relu(anchor_g @ W1[g,tgt] + b1)            [ng x 4096] bf16
    gemm_bf16_kernel<true, 0><<<dim3(INTER / 128, BSZ / 128, LNG), 256, 0, stream>>>(
        anchor, nullptr, order, goff, gcnt, W1, b1, nullptr, Hb, HID, INTER, 0);
    // T = H @ W2[g,tgt] + b2                         [ng x 1024] f32 + bf16
    gemm_bf16_kernel<false, 1><<<dim3(HID / 128, BSZ / 128, LNG), 256, 0, stream>>>(
        nullptr, Hb, order, goff, gcnt, W2, b2, Tf, Tb, INTER, HID, 0);
    // HC = relu(T @ W1[tgt,g] + b1c)                 [ng x 4096] bf16 (reuse Hb)
    gemm_bf16_kernel<false, 0><<<dim3(INTER / 128, BSZ / 128, LNG), 256, 0, stream>>>(
        nullptr, Tb, order, goff, gcnt, W1, b1, nullptr, Hb, HID, INTER, 1);
    // C = HC @ W2[tgt,g] + b2c                       [ng x 1024] f32
    gemm_bf16_kernel<false, 1><<<dim3(HID / 128, BSZ / 128, LNG), 256, 0, stream>>>(
        nullptr, Hb, order, goff, gcnt, W2, b2, Cf, nullptr, INTER, HID, 1);

    epilogue_kernel<<<BSZ, 256, 0, stream>>>(anchor, Tf, Cf, order, Wd, bd, labels, out);
}
